// MoELayer_44985487458590
// MI455X (gfx1250) — matrix operand order repaired and checked by reading, not compiled, not run on hardware
//
#include <hip/hip_runtime.h>
#include <math.h>

#define D_MODEL 1024
#define D_FF    4096
#define NEXP    8
#define TOKENS  4096
#define TM      16      // tokens per FFN block (one WMMA M-tile)
#define CHUNK   2048    // columns of H staged in LDS per pass

typedef __attribute__((ext_vector_type(16))) __bf16 v16bf;
typedef __attribute__((ext_vector_type(8)))  float  v8f;
typedef unsigned int   u32;
typedef unsigned short u16;
typedef __attribute__((ext_vector_type(4))) u32 u32x4;

union BFrag {
  v16bf v;
  u16   h[16];
  u32x4 q[2];
};

// fp32 -> bf16 round-to-nearest-even
__device__ __forceinline__ u16 f2bf(float f) {
  u32 u = __builtin_bit_cast(u32, f);
  u32 r = u + 0x7FFFu + ((u >> 16) & 1u);
  return (u16)(r >> 16);
}

// A fragment (16x32 bf16, MxK) from an LDS row-major tile.
// lanes 0-15 (M=lane): elems 0-15 = K k0..k0+15
// lanes 16-31 (M=lane-16): elems 0-15 = K k0+16..k0+31
__device__ __forceinline__ v16bf load_a_lds(const u16* base, int lane, int k0, int stride) {
  int half = lane >> 4;
  int m    = lane & 15;
  const u16* row = base + m * stride;
  BFrag f;
  f.q[0] = *(const u32x4*)(row + k0 + half * 16);
  f.q[1] = *(const u32x4*)(row + k0 + half * 16 + 8);
  return f.v;
}

// Fallback B fragment (32x16 bf16, KxN) from row-major fp32 weights [K x ldw].
__device__ __forceinline__ v16bf load_b_global(const float* __restrict__ w, int ldw,
                                               int k0, int n0, int lane) {
  int col = n0 + (lane & 15);
  int kb  = k0 + ((lane >> 4) << 4);
  const float* p = w + (size_t)kb * ldw + col;
  BFrag f;
#pragma unroll
  for (int j = 0; j < 16; ++j) f.h[j] = f2bf(p[(size_t)j * ldw]);
  return f.v;
}

// Fast B fragment from pre-swizzled bf16 image: [nt][kt][lane][16] -> wave reads 1KB contiguous
__device__ __forceinline__ v16bf load_b_swz(const u16* __restrict__ wsw, int ktiles,
                                            int k0, int n0, int lane) {
  size_t off = (((size_t)(n0 >> 4) * ktiles + (k0 >> 5)) * 32 + lane) * 16;
  const u16* p = wsw + off;
  BFrag f;
  f.q[0] = *(const u32x4*)(p);
  f.q[1] = *(const u32x4*)(p + 8);
  return f.v;
}

// ---------------- Weight convert + swizzle (once per launch) ----------------
__device__ __forceinline__ void swz_one(const float* __restrict__ w, u16* __restrict__ dst,
                                        int K, int N, int e, unsigned idx) {
  int lane   = idx & 31;
  unsigned tileid = idx >> 5;            // nt*ktiles + kt
  int ktiles = K >> 5;
  int kt = tileid % ktiles;
  int nt = tileid / ktiles;
  int col = (nt << 4) + (lane & 15);
  int kb  = (kt << 5) + ((lane >> 4) << 4);
  const float* p = w + (size_t)e * K * N + (size_t)kb * N + col;
  u16* d = dst + (size_t)e * K * N + ((size_t)tileid * 32 + lane) * 16;
  BFrag f;
#pragma unroll
  for (int j = 0; j < 16; ++j) f.h[j] = f2bf(p[(size_t)j * N]);
  *(u32x4*)(d)     = f.q[0];
  *(u32x4*)(d + 8) = f.q[1];
}

__global__ void __launch_bounds__(256) moe_swizzle(
    const float* __restrict__ w1, const float* __restrict__ w2,
    u16* __restrict__ w1s, u16* __restrict__ w2s)
{
  const unsigned PER = (D_MODEL / 32) * (D_FF / 16) * 32;   // 262144 (same for w2)
  unsigned gid = blockIdx.x * 256 + threadIdx.x;            // < 2*NEXP*PER
  unsigned em  = gid / PER;                                 // 0..7 -> w1, 8..15 -> w2
  unsigned idx = gid % PER;
  int e = em & 7;
  if (em < NEXP) swz_one(w1, w1s, D_MODEL, D_FF, e, idx);
  else           swz_one(w2, w2s, D_FF, D_MODEL, e, idx);
}

// ---------------- Router: logits, noisy top-2, sparse softmax ----------------
__global__ void __launch_bounds__(256) moe_router(
    const float* __restrict__ x, const float* __restrict__ noise,
    const float* __restrict__ rw, const float* __restrict__ rb,
    const float* __restrict__ nw, const float* __restrict__ nb,
    float* __restrict__ gate_out, int* __restrict__ topk_idx, float* __restrict__ topk_gate)
{
  int t   = blockIdx.x;
  int tid = threadIdx.x;
  const float* xr = x + (size_t)t * D_MODEL;
  float ar[NEXP] = {}, an[NEXP] = {};
#pragma unroll
  for (int i = 0; i < D_MODEL / 256; ++i) {
    int d = tid + i * 256;
    float xv = xr[d];
    const float* rp = rw + d * NEXP;
    const float* np = nw + d * NEXP;
#pragma unroll
    for (int e = 0; e < NEXP; ++e) {
      ar[e] = fmaf(xv, rp[e], ar[e]);
      an[e] = fmaf(xv, np[e], an[e]);
    }
  }
#pragma unroll
  for (int off = 16; off > 0; off >>= 1)
#pragma unroll
    for (int e = 0; e < NEXP; ++e) {
      ar[e] += __shfl_down(ar[e], off, 32);
      an[e] += __shfl_down(an[e], off, 32);
    }
  __shared__ float sred[8][2 * NEXP];
  int wv = tid >> 5, lane = tid & 31;
  if (lane == 0) {
#pragma unroll
    for (int e = 0; e < NEXP; ++e) { sred[wv][e] = ar[e]; sred[wv][NEXP + e] = an[e]; }
  }
  __syncthreads();
  if (tid == 0) {
    float noisy[NEXP];
#pragma unroll
    for (int e = 0; e < NEXP; ++e) {
      float l = rb[e], s = nb[e];
      for (int w = 0; w < 8; ++w) { l += sred[w][e]; s += sred[w][NEXP + e]; }
      float sp = fmaxf(s, 0.0f) + log1pf(expf(-fabsf(s)));   // stable softplus
      noisy[e] = l + noise[(size_t)t * NEXP + e] * sp;
    }
    int i1 = 0; float v1 = noisy[0];
#pragma unroll
    for (int e = 1; e < NEXP; ++e) if (noisy[e] > v1) { v1 = noisy[e]; i1 = e; }
    int i2 = -1; float v2 = -INFINITY;
#pragma unroll
    for (int e = 0; e < NEXP; ++e) if (e != i1 && noisy[e] > v2) { v2 = noisy[e]; i2 = e; }
    float e2 = expf(v2 - v1);
    float p1 = 1.0f / (1.0f + e2);
    float p2 = e2 / (1.0f + e2);
    float* g = gate_out + (size_t)t * NEXP;
#pragma unroll
    for (int e = 0; e < NEXP; ++e) g[e] = 0.0f;
    g[i1] = p1; g[i2] = p2;
    topk_idx[t * 2]     = i1; topk_idx[t * 2 + 1]  = i2;
    topk_gate[t * 2]    = p1; topk_gate[t * 2 + 1] = p2;
  }
}

// ------------- Deterministic per-expert token lists (wave ballot scan) -------------
__global__ void __launch_bounds__(256) moe_build_lists(
    const int* __restrict__ topk_idx, const float* __restrict__ topk_gate,
    int* __restrict__ cnt, int* __restrict__ tok_list, float* __restrict__ gate_list)
{
  int e    = threadIdx.x >> 5;   // one wave per expert
  int lane = threadIdx.x & 31;
  int c = 0;
  for (int base = 0; base < TOKENS; base += 32) {
    int t = base + lane;
    int ia = topk_idx[t * 2], ib = topk_idx[t * 2 + 1];
    bool hit = (ia == e) || (ib == e);
    float g  = (ia == e) ? topk_gate[t * 2] : topk_gate[t * 2 + 1];
    u32 m = (u32)__ballot(hit);
    if (hit) {
      int pos = c + __builtin_popcount(m & ((1u << lane) - 1u));
      tok_list[e * TOKENS + pos]  = t;
      gate_list[e * TOKENS + pos] = g;
    }
    c += __builtin_popcount(m);
  }
  if (lane == 0) cnt[e] = c;
}

// ---------------- Fused expert FFN: relu(x@w1+b1)@w2+b2, gated scatter-add ----------------
template <bool SWZ>
__global__ void __launch_bounds__(256) moe_ffn(
    const float* __restrict__ x,
    const float* __restrict__ w1, const float* __restrict__ b1,
    const float* __restrict__ w2, const float* __restrict__ b2,
    const u16* __restrict__ w1s, const u16* __restrict__ w2s,
    const int* __restrict__ cnt, const int* __restrict__ tok_list,
    const float* __restrict__ gate_list, float* __restrict__ out)
{
  extern __shared__ char smem[];
  u16*   xl = (u16*)smem;                                        // [TM][D_MODEL] bf16 (32KB)
  u16*   hl = (u16*)(smem + TM * D_MODEL * 2);                   // [TM][CHUNK]  bf16 (64KB)
  float* gl = (float*)(smem + TM * D_MODEL * 2 + TM * CHUNK * 2);
  int*   tl = (int*)(gl + TM);

  int e    = blockIdx.x >> 8;
  int tile = blockIdx.x & 255;
  int cntE = cnt[e];
  if (tile * TM >= cntE) return;            // uniform early exit

  int tid = threadIdx.x, wv = tid >> 5, lane = tid & 31;

  if (tid < TM) {
    int row = tile * TM + tid;
    int tok = 0; float g = 0.0f;            // pad rows: token 0 with zero gate
    if (row < cntE) { tok = tok_list[e * TOKENS + row]; g = gate_list[e * TOKENS + row]; }
    tl[tid] = tok; gl[tid] = g;
  }
  __syncthreads();

  // stage x tile (gathered rows) as bf16 in LDS, coalesced
  for (int it = 0; it < (TM * D_MODEL) / 256; ++it) {
    int idx = it * 256 + tid;
    int m = idx >> 10, d = idx & (D_MODEL - 1);
    xl[idx] = f2bf(x[(size_t)tl[m] * D_MODEL + d]);
  }
  __syncthreads();

  const float* w1e  = w1  + (size_t)e * D_MODEL * D_FF;
  const float* w2e  = w2  + (size_t)e * D_FF * D_MODEL;
  const u16*   w1se = w1s + (size_t)e * D_MODEL * D_FF;
  const u16*   w2se = w2s + (size_t)e * D_FF * D_MODEL;
  const float* b1e  = b1  + (size_t)e * D_FF;
  const float* b2e  = b2  + (size_t)e * D_MODEL;

  v8f acc[8];
  v8f vzero = {};
#pragma unroll
  for (int j = 0; j < 8; ++j) acc[j] = vzero;

  for (int hc = 0; hc < D_FF / CHUNK; ++hc) {
    // -------- phase 1: h_chunk = relu(x @ w1 + b1), each wave owns 256 columns --------
    for (int nt = 0; nt < CHUNK / (8 * 16); ++nt) {
      int nloc = wv * (CHUNK / 8) + nt * 16;      // column base within chunk
      int ng   = hc * CHUNK + nloc;               // global H column base
      v8f c = {};
      for (int k0 = 0; k0 < D_MODEL; k0 += 32) {
        v16bf a = load_a_lds(xl, lane, k0, D_MODEL);
        v16bf b;
        if constexpr (SWZ) b = load_b_swz(w1se, D_MODEL / 32, k0, ng, lane);
        else               b = load_b_global(w1e, D_FF, k0, ng, lane);
        c = __builtin_amdgcn_wmma_f32_16x16x32_bf16(false, a, false, b, (short)0, c, false, false);
      }
      float bias = b1e[ng + (lane & 15)];
      int cl = nloc + (lane & 15);
#pragma unroll
      for (int i = 0; i < 8; ++i) {
        int m = i + ((lane >> 4) << 3);           // C/D layout: VGPR i -> M=i (+8 upper half)
        hl[m * CHUNK + cl] = f2bf(fmaxf(c[i] + bias, 0.0f));
      }
    }
    __syncthreads();
    // -------- phase 2: y += h_chunk @ w2_chunk, each wave owns 128 output columns --------
    for (int k0 = 0; k0 < CHUNK; k0 += 32) {
      v16bf a = load_a_lds(hl, lane, k0, CHUNK);
#pragma unroll
      for (int j = 0; j < 8; ++j) {
        int n0 = wv * 128 + j * 16;
        v16bf b;
        if constexpr (SWZ) b = load_b_swz(w2se, D_FF / 32, hc * CHUNK + k0, n0, lane);
        else               b = load_b_global(w2e, D_MODEL, hc * CHUNK + k0, n0, lane);
        acc[j] = __builtin_amdgcn_wmma_f32_16x16x32_bf16(false, a, false, b, (short)0, acc[j], false, false);
      }
    }
    __syncthreads();
  }

  // -------- epilogue: add b2, gate-scale, scatter-add into combined output --------
#pragma unroll
  for (int j = 0; j < 8; ++j) {
    int col = wv * 128 + j * 16 + (lane & 15);
    float bias = b2e[col];
#pragma unroll
    for (int i = 0; i < 8; ++i) {
      int m = i + ((lane >> 4) << 3);
      float v = (acc[j][i] + bias) * gl[m];
      unsafeAtomicAdd(&out[(size_t)tl[m] * D_MODEL + col], v);
    }
  }
}

extern "C" void kernel_launch(void* const* d_in, const int* in_sizes, int n_in,
                              void* d_out, int out_size, void* d_ws, size_t ws_size,
                              hipStream_t stream) {
  const float* x   = (const float*)d_in[0];
  const float* noi = (const float*)d_in[1];
  const float* rw  = (const float*)d_in[2];
  const float* rb  = (const float*)d_in[3];
  const float* nw  = (const float*)d_in[4];
  const float* nb  = (const float*)d_in[5];
  const float* w1  = (const float*)d_in[6];
  const float* b1  = (const float*)d_in[7];
  const float* w2  = (const float*)d_in[8];
  const float* b2  = (const float*)d_in[9];

  float* out      = (float*)d_out;                       // [T*D] combined, then [T*E] gate
  float* gate_out = out + (size_t)TOKENS * D_MODEL;

  char* ws = (char*)d_ws;
  size_t off = 0;
  int*   topk_idx  = (int*)(ws + off);   off += (size_t)TOKENS * 2 * sizeof(int);
  float* topk_gate = (float*)(ws + off); off += (size_t)TOKENS * 2 * sizeof(float);
  int*   cnt       = (int*)(ws + off);   off += 64;
  int*   tok_list  = (int*)(ws + off);   off += (size_t)NEXP * TOKENS * sizeof(int);
  float* gate_list = (float*)(ws + off); off += (size_t)NEXP * TOKENS * sizeof(float);
  u16*   w1s       = (u16*)(ws + off);   off += (size_t)NEXP * D_MODEL * D_FF * sizeof(u16);
  u16*   w2s       = (u16*)(ws + off);   off += (size_t)NEXP * D_FF * D_MODEL * sizeof(u16);
  bool useSwz = (ws_size >= off);

  hipMemsetAsync(d_out, 0, (size_t)out_size * sizeof(float), stream);
  moe_router<<<TOKENS, 256, 0, stream>>>(x, noi, rw, rb, nw, nb, gate_out, topk_idx, topk_gate);
  moe_build_lists<<<1, 256, 0, stream>>>(topk_idx, topk_gate, cnt, tok_list, gate_list);

  size_t lds = (size_t)TM * D_MODEL * 2 + (size_t)TM * CHUNK * 2
             + TM * sizeof(float) + TM * sizeof(int);    // 98,432 B (<320KB WGP LDS)

  if (useSwz) {
    unsigned per   = (D_MODEL / 32) * (D_FF / 16) * 32;              // 262144
    unsigned total = 2u * NEXP * per;                                // 4,194,304
    moe_swizzle<<<total / 256, 256, 0, stream>>>(w1, w2, w1s, w2s);
    moe_ffn<true><<<NEXP * 256, 256, lds, stream>>>(x, w1, b1, w2, b2, w1s, w2s,
                                                    cnt, tok_list, gate_list, out);
  } else {
    moe_ffn<false><<<NEXP * 256, 256, lds, stream>>>(x, w1, b1, w2, b2, w1s, w2s,
                                                     cnt, tok_list, gate_list, out);
  }
}
